// AutoCompleteDecoderModel_77171972374637
// MI455X (gfx1250) — matrix operand order repaired and checked by reading, not compiled, not run on hardware
//
#include <hip/hip_runtime.h>

// ---------------------------------------------------------------------------
// Seq2seq LSTM (128 vocab, H=1024, B=128, 512 enc + 512 dec steps) for
// MI455X / gfx1250.  Persistent kernel, bf16 WMMA 16x16x32, one-hot input
// folded into the K-loop (K = 1024 + 128), device-wide barrier per timestep.
// K-loop is software-pipelined (2-deep ping-pong) to overlap global loads
// with WMMA issue.  Workspace required: ~20.3 MB.
// ---------------------------------------------------------------------------

typedef __attribute__((ext_vector_type(16))) __bf16 v16bf;
typedef __attribute__((ext_vector_type(8)))  __bf16 v8bf;
typedef __attribute__((ext_vector_type(8)))  float  v8f;

#define A_   128      // vocab
#define H_   1024     // hidden
#define B_   128      // batch
#define TC_  512      // encoder steps
#define TT_  513      // decoder tokens (512 input steps)
#define G4_  4096     // 4*H
#define KC_  1152     // H + A  (concatenated K)
#define NBLK 64
#define TPB  256

union AFrag { v16bf v; v8bf h[2]; __bf16 e[16]; };

__device__ __forceinline__ float sigm(float x) { return 1.0f / (1.0f + __expf(-x)); }

__device__ __forceinline__ void load_a(AFrag& a, const __bf16* ha, int kb, int lh) {
  const int e0 = kb * 32 + lh * 16;
  a.h[0] = *(const v8bf*)(ha + e0);
  a.h[1] = *(const v8bf*)(ha + e0 + 8);
}

__device__ __forceinline__ void load_b4(v16bf b[4], const __bf16* const wb[4], int kb, int lh) {
  const int bo = kb * 32 + lh * 16;
#pragma unroll
  for (int g = 0; g < 4; ++g) b[g] = *(const v16bf*)(wb[g] + bo);
}

__device__ __forceinline__ void wmma4(v8f acc[4], const AFrag& a, const v16bf b[4]) {
#pragma unroll
  for (int g = 0; g < 4; ++g)
    acc[g] = __builtin_amdgcn_wmma_f32_16x16x32_bf16(
        false, a.v, false, b[g], (short)0, acc[g], false, false);
}

// one-hot A fragment for K-block kb (kb in [32,36)): onehot cols = K-1024
__device__ __forceinline__ void build_onehot(AFrag& a, int kb, int lh, int myidx) {
  const int k0 = kb * 32 + lh * 16 - H_;
#pragma unroll
  for (int q = 0; q < 8; ++q) {
    a.e[q]     = (k0 + q      == myidx) ? (__bf16)1.0f : (__bf16)0.0f;
    a.e[q + 8] = (k0 + 8 + q  == myidx) ? (__bf16)1.0f : (__bf16)0.0f;
  }
}

// ---- weight packing: Wcat[n][k] = k<H ? W_hh[n][k] : W_ih[n][k-H], bf16 ----
__global__ void pack_cat_kernel(__bf16* __restrict__ dst,
                                const float* __restrict__ whh,
                                const float* __restrict__ wih) {
  int i = blockIdx.x * blockDim.x + threadIdx.x;
  if (i >= G4_ * KC_) return;
  int n = i / KC_, k = i % KC_;
  float v = (k < H_) ? whh[(size_t)n * H_ + k] : wih[(size_t)n * A_ + (k - H_)];
  dst[i] = (__bf16)v;
}

__global__ void pack_proj_kernel(__bf16* __restrict__ dst, const float* __restrict__ w) {
  int i = blockIdx.x * blockDim.x + threadIdx.x;
  if (i >= A_ * H_) return;
  dst[i] = (__bf16)w[i];
}

// ---- zero-init of recurrent state / accumulators / sync words (every call) --
__global__ void init_kernel(__bf16* hbf, float* cbuf, float* accum, unsigned* sync) {
  int i = blockIdx.x * blockDim.x + threadIdx.x;
  if (i < B_ * H_) {
    cbuf[i] = 0.0f;
    hbf[i] = (__bf16)0.0f;
    hbf[i + B_ * H_] = (__bf16)0.0f;
  }
  if (i == 0) { accum[0] = 0.0f; accum[1] = 0.0f; sync[0] = 0u; sync[1] = 0u; }
}

// ---- device-wide generation barrier (all 64 blocks co-resident) ------------
__device__ __forceinline__ void grid_sync(unsigned* cnt, unsigned* gen, unsigned nb) {
  __syncthreads();
  __threadfence();
  if (threadIdx.x == 0) {
    volatile unsigned* vgen = (volatile unsigned*)gen;
    const unsigned g = *vgen;
    if (atomicAdd(cnt, 1u) == nb - 1u) {
      atomicExch(cnt, 0u);
      __threadfence();
      atomicAdd(gen, 1u);
    } else {
      while (*vgen == g) __builtin_amdgcn_s_sleep(1);
    }
  }
  __syncthreads();
  __threadfence();
}

__global__ void __launch_bounds__(TPB, 1)
seq2seq_kernel(const int* __restrict__ src, const int* __restrict__ tgt,
               const float* __restrict__ enc_b, const float* __restrict__ dec_b,
               const float* __restrict__ proj_b,
               const __bf16* __restrict__ wcat_e, const __bf16* __restrict__ wcat_d,
               const __bf16* __restrict__ wproj,
               __bf16* __restrict__ hbf, float* __restrict__ cbuf,
               float* __restrict__ logits, float* __restrict__ accum,
               unsigned* sync, float* __restrict__ out) {
  __shared__ float s_nll[B_];
  __shared__ float s_msk[B_];

  const int tid   = threadIdx.x;
  const int wave  = tid >> 5;         // 0..7   -> batch tile (M)
  const int lane  = tid & 31;
  const int lr    = lane & 15;
  const int lh    = lane >> 4;        // half-wave select
  const int jtile = blockIdx.x;       // 0..63  -> 16 hidden units (N slice)
  const size_t HB = (size_t)B_ * H_;

  // Pipeline: phase s does LSTM step s, projection for decode step s-513,
  // softmax/NLL for decode step s-514.  1024 LSTM phases + 2 drain phases.
  for (int s = 0; s < 1026; ++s) {
    // ---------------- LSTM gate GEMM + cell update ----------------
    if (s < 1024) {
      const bool enc       = s < 512;
      const int  t         = enc ? s : s - 512;
      const int* idx       = enc ? src : tgt;
      const int  istr      = enc ? TC_ : TT_;
      const __bf16* wcat   = enc ? wcat_e : wcat_d;
      const float* bias    = enc ? enc_b : dec_b;
      const __bf16* hin    = hbf + (size_t)(s & 1) * HB;
      __bf16*       hout   = hbf + (size_t)((s + 1) & 1) * HB;

      const int arow  = wave * 16 + lr;                 // A-matrix row this lane feeds
      const int myidx = idx[arow * istr + t];           // token for one-hot K-tail
      const __bf16* ha = hin + (size_t)arow * H_;

      v8f acc[4];
      const __bf16* wb[4];
#pragma unroll
      for (int g = 0; g < 4; ++g) {
        const float bv = bias[g * H_ + jtile * 16 + lr];
#pragma unroll
        for (int r = 0; r < 8; ++r) acc[g][r] = bv;
        wb[g] = wcat + (size_t)(g * H_ + jtile * 16 + lr) * KC_;
        // warm L0 for the one-hot tail columns (K = 1024..1151)
        __builtin_prefetch(wb[g] + H_ + lh * 16, 0, 3);
        __builtin_prefetch(wb[g] + H_ + 64 + lh * 16, 0, 3);
      }

      // ---- main K loop (h part, K=1024), 2-deep software pipeline ----
      AFrag a0, a1;
      v16bf b0[4], b1[4];
      load_a(a0, ha, 0, lh);
      load_b4(b0, wb, 0, lh);
      for (int kb = 0; kb < 32; kb += 2) {
        load_a(a1, ha, kb + 1, lh);
        load_b4(b1, wb, kb + 1, lh);
        wmma4(acc, a0, b0);                 // consumes kb, while kb+1 in flight
        if (kb + 2 < 32) {
          load_a(a0, ha, kb + 2, lh);
          load_b4(b0, wb, kb + 2, lh);
        }
        wmma4(acc, a1, b1);                 // consumes kb+1, while kb+2 in flight
      }

      // ---- one-hot tail (K = 1024..1151), fully unrolled + pipelined ----
      load_b4(b0, wb, 32, lh);
#pragma unroll
      for (int kb = 32; kb < 36; ++kb) {
        if (kb + 1 < 36) {
          if ((kb + 1) & 1) load_b4(b1, wb, kb + 1, lh);
          else              load_b4(b0, wb, kb + 1, lh);
        }
        AFrag a;
        build_onehot(a, kb, lh, myidx);
        if (kb & 1) wmma4(acc, a, b1);
        else        wmma4(acc, a, b0);
      }

      // ---- elementwise cell update ----
      const int col = jtile * 16 + lr;
#pragma unroll
      for (int r = 0; r < 8; ++r) {
        const int row  = wave * 16 + lh * 8 + r;        // C/D layout: M = r + 8*lh
        const size_t o = (size_t)row * H_ + col;
        const float iv = acc[0][r], fv = acc[1][r], gv = acc[2][r], ov = acc[3][r];
        const float cn = sigm(fv) * cbuf[o] + sigm(iv) * tanhf(gv);
        cbuf[o] = cn;
        hout[o] = (__bf16)(sigm(ov) * tanhf(cn));
      }
    }

    // ---------------- projection for decode step d = s-513 ----------------
    const int d = s - 513;
    if (d >= 0 && d < 512 && wave == (blockIdx.x & 7)) {
      const int mt = blockIdx.x >> 3, nt = blockIdx.x & 7;   // 8x8 tiles of 128x128
      const __bf16* hp = hbf + (size_t)(s & 1) * HB + (size_t)(mt * 16 + lr) * H_;
      const __bf16* wp = wproj + (size_t)(nt * 16 + lr) * H_;
      v8f pacc;
      const float pb = proj_b[nt * 16 + lr];
#pragma unroll
      for (int r = 0; r < 8; ++r) pacc[r] = pb;

      AFrag a0, a1;
      v16bf w0, w1;
      load_a(a0, hp, 0, lh);
      w0 = *(const v16bf*)(wp + lh * 16);
      for (int kb = 0; kb < 32; kb += 2) {
        load_a(a1, hp, kb + 1, lh);
        w1 = *(const v16bf*)(wp + (kb + 1) * 32 + lh * 16);
        pacc = __builtin_amdgcn_wmma_f32_16x16x32_bf16(
            false, a0.v, false, w0, (short)0, pacc, false, false);
        if (kb + 2 < 32) {
          load_a(a0, hp, kb + 2, lh);
          w0 = *(const v16bf*)(wp + (kb + 2) * 32 + lh * 16);
        }
        pacc = __builtin_amdgcn_wmma_f32_16x16x32_bf16(
            false, a1.v, false, w1, (short)0, pacc, false, false);
      }

      float* lbuf = logits + (size_t)(d & 1) * (B_ * A_);
#pragma unroll
      for (int r = 0; r < 8; ++r) {
        const int row = mt * 16 + lh * 8 + r;
        lbuf[row * A_ + nt * 16 + lr] = pacc[r];
      }
    }

    // ---------------- log-softmax + NLL for decode step e = s-514 ----------
    const int e = s - 514;
    if (e >= 0 && e < 512 && blockIdx.x == 0) {
      if (tid < B_) {
        const float* Lr = logits + (size_t)(e & 1) * (B_ * A_) + (size_t)tid * A_;
        float mx = -3.4e38f;
        for (int j = 0; j < A_; ++j) mx = fmaxf(mx, Lr[j]);
        float se = 0.0f;
        for (int j = 0; j < A_; ++j) se += __expf(Lr[j] - mx);
        const int   tg = tgt[tid * TT_ + e + 1];
        const float m  = (tg != 0) ? 1.0f : 0.0f;       // PAD == 0
        s_nll[tid] = (__logf(se) + mx - Lr[tg]) * m;
        s_msk[tid] = m;
      }
      __syncthreads();
      if (tid == 0) {
        float sn = 0.0f, sm = 0.0f;
        for (int j = 0; j < B_; ++j) { sn += s_nll[j]; sm += s_msk[j]; }
        accum[0] += sn;
        accum[1] += sm;
      }
    }

    grid_sync(sync, sync + 1, (unsigned)gridDim.x);
  }

  if (blockIdx.x == 0 && tid == 0) out[0] = accum[0] / fmaxf(accum[1], 1.0f);
}

// ---------------------------------------------------------------------------
extern "C" void kernel_launch(void* const* d_in, const int* in_sizes, int n_in,
                              void* d_out, int out_size, void* d_ws, size_t ws_size,
                              hipStream_t stream) {
  (void)in_sizes; (void)n_in; (void)out_size; (void)ws_size;
  const int*   src  = (const int*)  d_in[0];
  const int*   tgt  = (const int*)  d_in[1];
  const float* eWih = (const float*)d_in[2];
  const float* eWhh = (const float*)d_in[3];
  const float* eb   = (const float*)d_in[4];
  const float* dWih = (const float*)d_in[5];
  const float* dWhh = (const float*)d_in[6];
  const float* db   = (const float*)d_in[7];
  const float* pW   = (const float*)d_in[8];
  const float* pb   = (const float*)d_in[9];

  // workspace layout (bf16 region first, then f32 region) — ~20.3 MB total
  const size_t NWCAT = (size_t)G4_ * KC_;
  __bf16*   wcat_e = (__bf16*)d_ws;
  __bf16*   wcat_d = wcat_e + NWCAT;
  __bf16*   wproj  = wcat_d + NWCAT;
  __bf16*   hbf    = wproj + (size_t)A_ * H_;             // 2 ping-pong h buffers
  float*    cbuf   = (float*)(hbf + 2 * (size_t)B_ * H_);
  float*    logits = cbuf + (size_t)B_ * H_;              // 2 ping-pong logit bufs
  float*    accum  = logits + 2 * (size_t)B_ * A_;        // [nll_sum, mask_sum]
  unsigned* sync   = (unsigned*)(accum + 2);              // [count, generation]

  pack_cat_kernel <<<(G4_ * KC_ + TPB - 1) / TPB, TPB, 0, stream>>>(wcat_e, eWhh, eWih);
  pack_cat_kernel <<<(G4_ * KC_ + TPB - 1) / TPB, TPB, 0, stream>>>(wcat_d, dWhh, dWih);
  pack_proj_kernel<<<(A_ * H_   + TPB - 1) / TPB, TPB, 0, stream>>>(wproj, pW);
  init_kernel     <<<(B_ * H_   + TPB - 1) / TPB, TPB, 0, stream>>>(hbf, cbuf, accum, sync);

  seq2seq_kernel<<<NBLK, TPB, 0, stream>>>(src, tgt, eb, db, pb,
                                           wcat_e, wcat_d, wproj,
                                           hbf, cbuf, logits, accum, sync,
                                           (float*)d_out);
}
